// PVConv_29892972380775
// MI455X (gfx1250) — compile-verified, matrix operand order checked by reading, not run on hardware
//
#include <hip/hip_runtime.h>
#include <hip/hip_bf16.h>

// ---------------------------------------------------------------------------
// PVConv block for MI455X (gfx1250, wave32). Convs run as implicit-im2col
// GEMMs on v_wmma_f32_16x16x32_bf16 with f32 accumulation.
// ---------------------------------------------------------------------------

typedef __attribute__((ext_vector_type(16))) __bf16 v16bf;
typedef __attribute__((ext_vector_type(8)))  __bf16 v8bf;
typedef __attribute__((ext_vector_type(8)))  float  v8f;

#define Bsz   8
#define Npts  4096
#define CIN   64
#define COUT  128
#define Rg    32
#define NPTS  (Bsz * Npts)            // 32768
#define NVOX  (Bsz * Rg * Rg * Rg)    // 262144

__device__ inline float pv_toF(float v) { return v; }
__device__ inline float pv_toF(__hip_bfloat16 v) { return __bfloat162float(v); }
__device__ inline void  pv_store(float& d, float v) { d = v; }
__device__ inline void  pv_store(__hip_bfloat16& d, float v) { d = __float2bfloat16(v); }

// ---------------- coords: per-batch mean + max radius ----------------------
__global__ void pv_coord_stats(const float* __restrict__ inputs, float* __restrict__ stats) {
    __shared__ float sx[256], sy[256], sz[256];
    int b = blockIdx.x, t = threadIdx.x;
    float ax = 0.f, ay = 0.f, az = 0.f;
    for (int n = t; n < Npts; n += 256) {
        const float* p = &inputs[(size_t)(b * Npts + n) * 67];
        ax += p[0]; ay += p[1]; az += p[2];
    }
    sx[t] = ax; sy[t] = ay; sz[t] = az;
    __syncthreads();
    for (int s = 128; s > 0; s >>= 1) {
        if (t < s) { sx[t] += sx[t + s]; sy[t] += sy[t + s]; sz[t] += sz[t + s]; }
        __syncthreads();
    }
    float mx = sx[0] / (float)Npts, my = sy[0] / (float)Npts, mz = sz[0] / (float)Npts;
    __syncthreads();
    float mr = 0.f;
    for (int n = t; n < Npts; n += 256) {
        const float* p = &inputs[(size_t)(b * Npts + n) * 67];
        float dx = p[0] - mx, dy = p[1] - my, dz = p[2] - mz;
        mr = fmaxf(mr, dx * dx + dy * dy + dz * dz);
    }
    sx[t] = mr;
    __syncthreads();
    for (int s = 128; s > 0; s >>= 1) {
        if (t < s) sx[t] = fmaxf(sx[t], sx[t + s]);
        __syncthreads();
    }
    if (t == 0) {
        stats[b * 4 + 0] = mx; stats[b * 4 + 1] = my; stats[b * 4 + 2] = mz;
        stats[b * 4 + 3] = sqrtf(sx[0]);
    }
}

// ---------------- voxelize: nc + voxel idx + counts ------------------------
__global__ void pv_voxelize_idx(const float* __restrict__ inputs, const float* __restrict__ stats,
                                float* __restrict__ nc, int* __restrict__ vidx, float* __restrict__ cnts) {
    int t = blockIdx.x * 256 + threadIdx.x;
    if (t >= NPTS) return;
    int b = t >> 12;
    const float* p  = &inputs[(size_t)t * 67];
    const float* st = &stats[b * 4];
    float inv = 1.f / (st[3] * 2.f);
    int vx[3];
#pragma unroll
    for (int k = 0; k < 3; ++k) {
        float v = (p[k] - st[k]) * inv + 0.5f;
        v = v * (float)Rg;
        v = fminf(fmaxf(v, 0.f), (float)(Rg - 1));   // clip(nc*R, 0, R-1)
        nc[(size_t)t * 3 + k] = v;
        vx[k] = (int)rintf(v);                       // round-half-to-even, as jnp.round
    }
    int idx = ((vx[0] * Rg + vx[1]) * Rg + vx[2]) + (b << 15);
    vidx[t] = idx;
    atomicAdd(&cnts[idx], 1.f);
}

// ---------------- scatter feats into voxel sums ----------------------------
__global__ void pv_voxelize_scatter(const float* __restrict__ inputs, const int* __restrict__ vidx,
                                    float* __restrict__ sums) {
    int t = blockIdx.x * 256 + threadIdx.x;   // exactly NPTS*64 threads
    int p = t >> 6, c = t & 63;
    atomicAdd(&sums[(size_t)vidx[p] * CIN + c], inputs[(size_t)p * 67 + 3 + c]);
}

// ---------------- average -> bf16 grid -------------------------------------
__global__ void pv_vox_avg(const float* __restrict__ sums, const float* __restrict__ cnts,
                           __hip_bfloat16* __restrict__ grid) {
    size_t i = (size_t)blockIdx.x * 256 + threadIdx.x;   // exactly NVOX*64
    float c = cnts[i >> 6];
    grid[i] = __float2bfloat16(sums[i] / fmaxf(c, 1.f));
}

// ---------------- repack conv weights into per-lane WMMA fragment order ----
// wswz linear index t = ((kc*8 + nt)*32 + lane)*16 + e  maps to (k, n):
//   khalf = (lane>=16)*8 ; kin = e<8 ? khalf+e : 16+khalf+(e-8) ; k = kc*32+kin
//   n = nt*16 + (lane&15) ; k = tap*CINK + cin ; tap = (dx,dy,dz)
__global__ void pv_repack_w(const float* __restrict__ w, __hip_bfloat16* __restrict__ wswz,
                            int CINK, int total) {
    int t = blockIdx.x * 256 + threadIdx.x;
    if (t >= total) return;
    int e = t & 15, lane = (t >> 4) & 31, nt = (t >> 9) & 7, kc = t >> 12;
    int khalf = (lane >> 4) * 8;
    int kin = (e < 8) ? (khalf + e) : (16 + khalf + (e - 8));
    int k = kc * 32 + kin;
    int n = nt * 16 + (lane & 15);
    int tap = k / CINK, cin = k % CINK;
    int dx = tap / 9, dy = (tap / 3) % 3, dz = tap % 3;
    wswz[t] = __float2bfloat16(w[((((size_t)n * CINK + cin) * 3 + dx) * 3 + dy) * 3 + dz]);
}

// ---------------- 3x3x3 conv as WMMA GEMM ----------------------------------
// Block: 16 z-rows (fixed b,x,y) x 128 couts; 8 waves, one 16x16 tile each.
template<int CINT>
__global__ __launch_bounds__(256) void pv_conv3d_wmma(
        const __hip_bfloat16* __restrict__ in,
        const __hip_bfloat16* __restrict__ wswz,
        __hip_bfloat16* __restrict__ out) {
    __shared__ alignas(16) __bf16 ABUF[3 * 3 * 18 * CINT];

    const int blk = blockIdx.x;
    const int zt  = blk & 1;
    const int y   = (blk >> 1) & 31;
    const int x   = (blk >> 6) & 31;
    const int b   = blk >> 11;
    const int tid = threadIdx.x;

    // Stage input halo (3x3x18 positions x CINT channels) into LDS, zero-padded.
    const int VPP  = CINT / 8;          // 16B vectors per position
    const int NVEC = 162 * VPP;
    for (int i = tid; i < NVEC; i += 256) {
        int pos = i / VPP, v = i - pos * VPP;
        int zz = pos % 18;
        int t2 = pos / 18;
        int ddy = t2 % 3, ddx = t2 / 3;
        int gx = x + ddx - 1, gy = y + ddy - 1, gz = zt * 16 + zz - 1;
        uint4 val = {0u, 0u, 0u, 0u};
        if ((unsigned)gx < 32u && (unsigned)gy < 32u && (unsigned)gz < 32u) {
            size_t off = ((size_t)((((b * 32) + gx) * 32 + gy) * 32 + gz)) * CINT + v * 8;
            val = *(const uint4*)(in + off);
        }
        *(uint4*)(&ABUF[pos * CINT + v * 8]) = val;
    }
    __syncthreads();

    const int lane   = tid & 31;
    const int wv     = tid >> 5;     // n-tile 0..7
    const int m      = lane & 15;    // A row
    const int laneHi = lane >> 4;
    const int khalf  = laneHi * 8;
    const int HC     = CINT / 32;    // k-chunks per tap

    v8f acc = {0.f, 0.f, 0.f, 0.f, 0.f, 0.f, 0.f, 0.f};
    const __bf16* wp = (const __bf16*)wswz;

    for (int tap = 0; tap < 27; ++tap) {
        const int dx = tap / 9, dy = (tap / 3) % 3, dz = tap % 3;
        const __bf16* abase = &ABUF[((dx * 3 + dy) * 18 + (m + dz)) * CINT + khalf];
#pragma unroll
        for (int hc = 0; hc < HC; ++hc) {
            v8bf lo = *(const v8bf*)(abase + hc * 32);        // cin [khalf, khalf+8)
            v8bf hi = *(const v8bf*)(abase + hc * 32 + 16);   // cin [16+khalf, ...)
            v16bf a;
#pragma unroll
            for (int q = 0; q < 8; ++q) { a[q] = lo[q]; a[8 + q] = hi[q]; }
            const int kc = tap * HC + hc;
            v16bf bfrag = *(const v16bf*)(wp + ((size_t)((kc * 8 + wv) * 32 + lane)) * 16);
            acc = __builtin_amdgcn_wmma_f32_16x16x32_bf16(false, a, false, bfrag,
                                                          (short)0, acc, false, false);
        }
    }

    // C/D layout: VGPR r -> M = r + 8*laneHi ; N = lane&15
    const int n = wv * 16 + (lane & 15);
#pragma unroll
    for (int r = 0; r < 8; ++r) {
        int z = zt * 16 + r + laneHi * 8;
        size_t vox = (size_t)((((b * 32) + x) * 32 + y) * 32 + z);
        out[vox * COUT + n] = __float2bfloat16(acc[r]);
    }
}

// ---------------- per-channel sum / sumsq (BN stats) -----------------------
template<typename T>
__global__ void pv_chanstats(const T* __restrict__ x, int rows,
                             float* __restrict__ sum, float* __restrict__ sumsq) {
    int c  = threadIdx.x & 127;
    int rl = threadIdx.x >> 7;                 // 0..1
    float s = 0.f, q = 0.f;
    for (int r = blockIdx.x * 2 + rl; r < rows; r += gridDim.x * 2) {
        float v = pv_toF(x[(size_t)r * 128 + c]);
        s += v; q += v * v;
    }
    atomicAdd(&sum[c], s);
    atomicAdd(&sumsq[c], q);
}

__global__ void pv_bn_finalize(const float* __restrict__ sum, const float* __restrict__ sq,
                               const float* __restrict__ g, const float* __restrict__ be,
                               float invcnt, float eps,
                               float* __restrict__ scale, float* __restrict__ shift) {
    int c = threadIdx.x;
    float m = sum[c] * invcnt;
    float v = sq[c] * invcnt - m * m;
    float s = g[c] * rsqrtf(v + eps);
    scale[c] = s;
    shift[c] = be[c] - m * s;
}

template<typename T>
__global__ void pv_bn_apply(T* __restrict__ x, const float* __restrict__ scale,
                            const float* __restrict__ shift, size_t total, float slope) {
    size_t stride = (size_t)gridDim.x * blockDim.x;
    for (size_t i = (size_t)blockIdx.x * blockDim.x + threadIdx.x; i < total; i += stride) {
        int c = (int)(i & 127);
        float v = pv_toF(x[i]) * scale[c] + shift[c];
        v = v > 0.f ? v : slope * v;
        pv_store(x[i], v);
    }
}

// ---------------- point branch: pf = W_point @ feats + b -------------------
__global__ void pv_pointwise(const float* __restrict__ inputs, const float* __restrict__ w_point,
                             const float* __restrict__ b_point, float* __restrict__ pf) {
    __shared__ float fl[CIN];
    int p = blockIdx.x, c = threadIdx.x;     // 128 threads
    if (c < CIN) fl[c] = inputs[(size_t)p * 67 + 3 + c];
    __syncthreads();
    float s = b_point[c];
#pragma unroll 8
    for (int k = 0; k < CIN; ++k) s += w_point[c * CIN + k] * fl[k];
    pf[(size_t)p * 128 + c] = s;
}

// ---------------- sum pf over points into fused ----------------------------
__global__ void pv_pfsum(const float* __restrict__ pf, float* __restrict__ fused) {
    int b = blockIdx.x >> 4, ch = blockIdx.x & 15;
    int c = threadIdx.x & 127, rl = threadIdx.x >> 7;
    float s = 0.f;
    for (int r = rl; r < 256; r += 2) {
        int n = ch * 256 + r;
        s += pf[((size_t)(b * Npts + n)) * 128 + c];
    }
    atomicAdd(&fused[b * 128 + c], s);
}

// ---------------- trilinear devoxelize + sum over points -------------------
__global__ void pv_devox(const __hip_bfloat16* __restrict__ v2, const float* __restrict__ nc,
                         float* __restrict__ fused) {
    int c = threadIdx.x;                 // channel 0..127
    int b = blockIdx.x >> 5;
    int pch = blockIdx.x & 31;
    size_t base = (size_t)b * (Rg * Rg * Rg);
    float total = 0.f;
    for (int pp = 0; pp < 128; ++pp) {
        int n = pch * 128 + pp;
        const float* p = &nc[((size_t)(b * Npts + n)) * 3];
        float fx = p[0], fy = p[1], fz = p[2];
        int lx = (int)floorf(fx), ly = (int)floorf(fy), lz = (int)floorf(fz);
        int hx = min(lx + 1, Rg - 1), hy = min(ly + 1, Rg - 1), hz = min(lz + 1, Rg - 1);
        float ax = fx - (float)lx, ay = fy - (float)ly, az = fz - (float)lz;
#pragma unroll
        for (int dx = 0; dx < 2; ++dx) {
            int X = dx ? hx : lx; float wx = dx ? ax : 1.f - ax;
#pragma unroll
            for (int dy = 0; dy < 2; ++dy) {
                int Y = dy ? hy : ly; float wy = dy ? ay : 1.f - ay;
#pragma unroll
                for (int dz = 0; dz < 2; ++dz) {
                    int Z = dz ? hz : lz; float wz = dz ? az : 1.f - az;
                    size_t idx = base + (size_t)((X * Rg + Y) * Rg + Z);
                    total += wx * wy * wz * __bfloat162float(v2[idx * COUT + c]);
                }
            }
        }
    }
    atomicAdd(&fused[b * 128 + c], total);
}

// ---------------- head: fc1 -> LN -> tanh -> fc2 ---------------------------
__global__ void pv_head(const float* __restrict__ fused, const float* __restrict__ w_fc1,
                        const float* __restrict__ b_fc1, const float* __restrict__ g_ln,
                        const float* __restrict__ be_ln, const float* __restrict__ w_fc2,
                        const float* __restrict__ b_fc2, float* __restrict__ outp) {
    __shared__ float red[128];
    int b = blockIdx.x, c = threadIdx.x;
    float h = b_fc1[c];
#pragma unroll 8
    for (int k = 0; k < 128; ++k) h += fused[b * 128 + k] * w_fc1[c * 128 + k];

    red[c] = h; __syncthreads();
    for (int s = 64; s > 0; s >>= 1) { if (c < s) red[c] += red[c + s]; __syncthreads(); }
    float mu = red[0] / 128.f; __syncthreads();
    float d = h - mu;
    red[c] = d * d; __syncthreads();
    for (int s = 64; s > 0; s >>= 1) { if (c < s) red[c] += red[c + s]; __syncthreads(); }
    float var = red[0] / 128.f; __syncthreads();
    float t = tanhf(d * rsqrtf(var + 1e-5f) * g_ln[c] + be_ln[c]);
    red[c] = t * w_fc2[c]; __syncthreads();
    for (int s = 64; s > 0; s >>= 1) { if (c < s) red[c] += red[c + s]; __syncthreads(); }
    if (c == 0) outp[b] = red[0] + b_fc2[0];
}

// ---------------- coords passthrough output --------------------------------
__global__ void pv_coords_out(const float* __restrict__ inputs, float* __restrict__ out) {
    int i = blockIdx.x * 256 + threadIdx.x;
    if (i >= Bsz * 3 * Npts) return;
    int n = i & 4095;
    int bk = i >> 12;            // = b*3 + k
    int k = bk % 3, b = bk / 3;
    out[i] = inputs[((size_t)(b * Npts + n)) * 67 + k];
}

// ===========================================================================
extern "C" void kernel_launch(void* const* d_in, const int* in_sizes, int n_in,
                              void* d_out, int out_size, void* d_ws, size_t ws_size,
                              hipStream_t stream) {
    const float* inputs  = (const float*)d_in[0];
    const float* w_conv1 = (const float*)d_in[1];
    // d_in[2] b_conv1, d_in[6] b_conv2: constant per-channel shifts, provably
    // cancelled by the following training-mode BatchNorms -> skipped.
    const float* g_bn1   = (const float*)d_in[3];
    const float* be_bn1  = (const float*)d_in[4];
    const float* w_conv2 = (const float*)d_in[5];
    const float* g_bn2   = (const float*)d_in[7];
    const float* be_bn2  = (const float*)d_in[8];
    const float* w_point = (const float*)d_in[9];
    const float* b_point = (const float*)d_in[10];
    const float* g_bnp   = (const float*)d_in[11];
    const float* be_bnp  = (const float*)d_in[12];
    const float* w_fc1   = (const float*)d_in[13];
    const float* b_fc1   = (const float*)d_in[14];
    const float* g_ln    = (const float*)d_in[15];
    const float* be_ln   = (const float*)d_in[16];
    const float* w_fc2   = (const float*)d_in[17];
    const float* b_fc2   = (const float*)d_in[18];
    float* out = (float*)d_out;

    // ---- deterministic linear workspace carve (~243 MB) ----
    char* wsp = (char*)d_ws;
    auto alloc = [&](size_t bytes) -> void* {
        void* p = (void*)wsp;
        wsp += (bytes + 255) & ~(size_t)255;
        return p;
    };
    float* stats = (float*)alloc(Bsz * 4 * sizeof(float));
    float* cstat = (float*)alloc(1536 * sizeof(float));
    float* fused = (float*)alloc(Bsz * 128 * sizeof(float));
    float* nc    = (float*)alloc((size_t)NPTS * 3 * sizeof(float));
    int*   vidx  = (int*)  alloc((size_t)NPTS * sizeof(int));
    float* cnts  = (float*)alloc((size_t)NVOX * sizeof(float));
    float* sums  = (float*)alloc((size_t)NVOX * CIN * sizeof(float));
    __hip_bfloat16* grid  = (__hip_bfloat16*)alloc((size_t)NVOX * CIN * 2);
    __hip_bfloat16* act1  = (__hip_bfloat16*)alloc((size_t)NVOX * COUT * 2);
    __hip_bfloat16* act2  = (__hip_bfloat16*)alloc((size_t)NVOX * COUT * 2);
    __hip_bfloat16* wswz1 = (__hip_bfloat16*)alloc((size_t)54 * 4096 * 2);
    __hip_bfloat16* wswz2 = (__hip_bfloat16*)alloc((size_t)108 * 4096 * 2);
    float* pf = (float*)alloc((size_t)NPTS * 128 * sizeof(float));
    (void)ws_size; (void)in_sizes; (void)n_in; (void)out_size;

    float* sum1 = cstat +   0; float* sq1 = cstat + 128;
    float* sum2 = cstat + 256; float* sq2 = cstat + 384;
    float* sump = cstat + 512; float* sqp = cstat + 640;
    float* s1 = cstat +  768; float* t1 = cstat +  896;
    float* s2 = cstat + 1024; float* t2 = cstat + 1152;
    float* sp = cstat + 1280; float* tp = cstat + 1408;

    hipMemsetAsync(cnts, 0, (size_t)NVOX * sizeof(float), stream);
    hipMemsetAsync(sums, 0, (size_t)NVOX * CIN * sizeof(float), stream);
    hipMemsetAsync(cstat, 0, 1536 * sizeof(float), stream);
    hipMemsetAsync(fused, 0, Bsz * 128 * sizeof(float), stream);

    // voxelization
    pv_coord_stats<<<Bsz, 256, 0, stream>>>(inputs, stats);
    pv_voxelize_idx<<<NPTS / 256, 256, 0, stream>>>(inputs, stats, nc, vidx, cnts);
    pv_voxelize_scatter<<<(NPTS * CIN) / 256, 256, 0, stream>>>(inputs, vidx, sums);
    pv_vox_avg<<<(NVOX * CIN) / 256, 256, 0, stream>>>(sums, cnts, grid);

    // weight repack into WMMA fragment order
    pv_repack_w<<<(54 * 4096 + 255) / 256, 256, 0, stream>>>(w_conv1, wswz1, CIN, 54 * 4096);
    pv_repack_w<<<(108 * 4096 + 255) / 256, 256, 0, stream>>>(w_conv2, wswz2, COUT, 108 * 4096);

    // conv1 -> BN -> leaky(0.1)
    pv_conv3d_wmma<CIN><<<16384, 256, 0, stream>>>(grid, wswz1, act1);
    pv_chanstats<__hip_bfloat16><<<1024, 256, 0, stream>>>(act1, NVOX, sum1, sq1);
    pv_bn_finalize<<<1, 128, 0, stream>>>(sum1, sq1, g_bn1, be_bn1, 1.f / NVOX, 1e-4f, s1, t1);
    pv_bn_apply<__hip_bfloat16><<<8192, 256, 0, stream>>>(act1, s1, t1, (size_t)NVOX * COUT, 0.1f);

    // conv2 -> BN -> leaky(0.1)
    pv_conv3d_wmma<COUT><<<16384, 256, 0, stream>>>(act1, wswz2, act2);
    pv_chanstats<__hip_bfloat16><<<1024, 256, 0, stream>>>(act2, NVOX, sum2, sq2);
    pv_bn_finalize<<<1, 128, 0, stream>>>(sum2, sq2, g_bn2, be_bn2, 1.f / NVOX, 1e-4f, s2, t2);
    pv_bn_apply<__hip_bfloat16><<<8192, 256, 0, stream>>>(act2, s2, t2, (size_t)NVOX * COUT, 0.1f);

    // point branch -> BN -> relu -> sum over points
    pv_pointwise<<<NPTS, 128, 0, stream>>>(inputs, w_point, b_point, pf);
    pv_chanstats<float><<<256, 256, 0, stream>>>(pf, NPTS, sump, sqp);
    pv_bn_finalize<<<1, 128, 0, stream>>>(sump, sqp, g_bnp, be_bnp, 1.f / NPTS, 1e-5f, sp, tp);
    pv_bn_apply<float><<<4096, 256, 0, stream>>>(pf, sp, tp, (size_t)NPTS * 128, 0.f);
    pv_pfsum<<<Bsz * 16, 256, 0, stream>>>(pf, fused);

    // devoxelize (trilinear gather) + sum over points into fused
    pv_devox<<<Bsz * 32, 128, 0, stream>>>(act2, nc, fused);

    // head -> pred, then coords passthrough
    pv_head<<<Bsz, 128, 0, stream>>>(fused, w_fc1, b_fc1, g_ln, be_ln, w_fc2, b_fc2, out);
    pv_coords_out<<<(Bsz * 3 * Npts + 255) / 256, 256, 0, stream>>>(inputs, out + Bsz);
}
